// RNNModel_58153857188079
// MI455X (gfx1250) — matrix operand order repaired and checked by reading, not compiled, not run on hardware
//
#include <hip/hip_runtime.h>

// ---------------- problem constants ----------------
#define T_STEPS 256
#define BS      64
#define NTOKEN  512
#define NINP    512
#define NHID    2048
#define NB      8
#define BSZ     256
#define TOPK    4
#define ATT_OUT 340
#define DK      64
#define NH_C    4
#define DK_C    32
#define VPAD    352              // ATT_OUT padded to mult of 32
#define KCAT    (VPAD + BSZ)     // 608, K of gates GEMM
#define QKV_N   (3*NH_C*DK_C)    // 384

// ---------------- WMMA plumbing (gfx1250, wave32) ----------------
typedef __attribute__((ext_vector_type(16))) __bf16 v16bf;
typedef __attribute__((ext_vector_type(8)))  __bf16 bf8v;
typedef __attribute__((ext_vector_type(8)))  float  v8f;

// A fragment: 16x32 bf16, row-major source. Lane l -> row l&15;
// lanes 0-15 hold K {k0..k0+7, k0+16..k0+23}, lanes 16-31 hold {+8, +24}.
__device__ __forceinline__ v16bf frag_a_bf16(const __bf16* A, int ld, int k0, int lane) {
  int r = lane & 15, h = lane >> 4;
  const __bf16* p = A + (size_t)r * ld + k0 + h * 8;
  bf8v lo = *(const bf8v*)p;
  bf8v hi = *(const bf8v*)(p + 16);
  return __builtin_shufflevector(lo, hi, 0,1,2,3,4,5,6,7,8,9,10,11,12,13,14,15);
}
// Same fragment but from an f32 row-major source (converted in registers).
__device__ __forceinline__ v16bf frag_a_f32(const float* A, int ld, int k0, int lane) {
  int r = lane & 15, h = lane >> 4;
  const float* p = A + (size_t)r * ld + k0 + h * 8;
  v16bf o;
#pragma unroll
  for (int i = 0; i < 8; ++i) { o[i] = (__bf16)p[i]; o[i + 8] = (__bf16)p[i + 16]; }
  return o;
}
// B fragment: 32x16 bf16 from B^T (N x K row-major). Lane l -> column l&15;
// lanes 0-15 hold K {k0..k0+7, k0+16..k0+23}, lanes 16-31 hold {+8, +24}.
__device__ __forceinline__ v16bf frag_b_bf16(const __bf16* BT, int ld, int k0, int lane) {
  int n = lane & 15, h = lane >> 4;
  const __bf16* p = BT + (size_t)n * ld + k0 + h * 8;
  bf8v lo = *(const bf8v*)p;
  bf8v hi = *(const bf8v*)(p + 16);
  return __builtin_shufflevector(lo, hi, 0,1,2,3,4,5,6,7,8,9,10,11,12,13,14,15);
}
#define WMMA_BF16(a, b, c) \
  __builtin_amdgcn_wmma_f32_16x16x32_bf16(false, (a), false, (b), (short)0, (c), false, false)

__device__ __forceinline__ float sigf(float x) { return 1.f / (1.f + __expf(-x)); }

// ---------------- setup kernels ----------------
// WcatT[n][col][k] bf16 : k<340 -> Wi[n][k][col]; 340..351 -> 0; >=352 -> Wh[n][k-352][col]
__global__ void kPrepWcat(const float* __restrict__ Wi, const float* __restrict__ Wh,
                          __bf16* __restrict__ WcatT) {
  int i = blockIdx.x * 256 + threadIdx.x;
  if (i >= NB * 1024 * KCAT) return;
  int n = i / (1024 * KCAT);
  int r = i - n * (1024 * KCAT);
  int col = r / KCAT;
  int k   = r - col * KCAT;
  float v;
  if (k < ATT_OUT)      v = Wi[((size_t)n * ATT_OUT + k) * 1024 + col];
  else if (k < VPAD)    v = 0.f;
  else                  v = Wh[((size_t)n * BSZ + (k - VPAD)) * 1024 + col];
  WcatT[i] = (__bf16)v;
}

// Transposed/padded bf16 (and one f32) weight panels for all other GEMMs.
__global__ void kPrepMisc(const float* __restrict__ W_enc, const float* __restrict__ Wk,
                          const float* __restrict__ Wv,    const float* __restrict__ Wq,
                          const float* __restrict__ Wq_c,  const float* __restrict__ Wk_c,
                          const float* __restrict__ Wv_c,
                          __bf16* __restrict__ WencT, __bf16* __restrict__ WkT,
                          __bf16* __restrict__ WvT,   float*  __restrict__ WqT,
                          __bf16* __restrict__ WqkvT) {
  int i = blockIdx.x * 256 + threadIdx.x;
  if (i < 512 * 512) { int c = i >> 9, k = i & 511; WencT[i] = (__bf16)W_enc[k * 512 + c]; return; }
  i -= 512 * 512;
  if (i < 64 * 512)  { int j = i >> 9, k = i & 511; WkT[i] = (__bf16)Wk[k * 64 + j]; return; }
  i -= 64 * 512;
  if (i < VPAD * 512) {
    int v = i >> 9, k = i & 511;
    WvT[i] = (__bf16)((v < ATT_OUT) ? Wv[k * ATT_OUT + v] : 0.f);
    return;
  }
  i -= VPAD * 512;
  if (i < 64 * 256)  { int j = i >> 8, k = i & 255; WqT[i] = Wq[k * 64 + j]; return; }
  i -= 64 * 256;
  if (i < QKV_N * 256) {
    int j = i >> 8, k = i & 255;
    float v = (j < 128) ? Wq_c[k * 128 + j]
            : (j < 256) ? Wk_c[k * 128 + (j - 128)]
                        : Wv_c[k * 128 + (j - 256)];
    WqkvT[i] = (__bf16)v;
    return;
  }
}

__global__ void kInitState(const float* __restrict__ h0, const float* __restrict__ c0,
                           float* __restrict__ hbuf, float* __restrict__ cbuf) {
  int i = blockIdx.x * 256 + threadIdx.x;
  if (i < BS * NHID) { hbuf[i] = h0[i]; cbuf[i] = c0[i]; }
}

// emb = x @ W_enc + b_enc   (M=16384, K=512, N=512), bf16 out.  One 16x16 tile/wave.
__global__ void kEmb(const float* __restrict__ X, const __bf16* __restrict__ WencT,
                     const float* __restrict__ b_enc, __bf16* __restrict__ emb) {
  int wid = blockIdx.x * 8 + (threadIdx.x >> 5);
  int lane = threadIdx.x & 31;
  int m0 = (wid >> 5) * 16, n0 = (wid & 31) * 16;
  const float*  A  = X + (size_t)m0 * NINP;
  const __bf16* BT = WencT + (size_t)n0 * NINP;
  v8f acc = {};
  for (int k0 = 0; k0 < NINP; k0 += 32) {
    v16bf a = frag_a_f32(A, NINP, k0, lane);
    v16bf b = frag_b_bf16(BT, NINP, k0, lane);
    acc = WMMA_BF16(a, b, acc);
  }
  int col = n0 + (lane & 15), rb = (lane >> 4) * 8;
  float bias = b_enc[col];
#pragma unroll
  for (int r = 0; r < 8; ++r)
    emb[(size_t)(m0 + rb + r) * NINP + col] = (__bf16)(acc[r] + bias);
}

// K_all (f32) and padded V_all (bf16) for all timesteps.  N tiles: 0..3 -> K, 4..25 -> V.
__global__ void kKV(const __bf16* __restrict__ emb, const __bf16* __restrict__ WkT,
                    const __bf16* __restrict__ WvT, float* __restrict__ Kall,
                    __bf16* __restrict__ Vall) {
  int wid = blockIdx.x * 8 + (threadIdx.x >> 5);
  int lane = threadIdx.x & 31;
  int mt = wid / 26, nt = wid - mt * 26;
  int m0 = mt * 16, n0 = nt * 16;
  const __bf16* A  = emb + (size_t)m0 * NINP;
  const __bf16* BT = (n0 < DK) ? WkT + (size_t)n0 * NINP : WvT + (size_t)(n0 - DK) * NINP;
  v8f acc = {};
  for (int k0 = 0; k0 < NINP; k0 += 32) {
    v16bf a = frag_a_bf16(A, NINP, k0, lane);
    v16bf b = frag_b_bf16(BT, NINP, k0, lane);
    acc = WMMA_BF16(a, b, acc);
  }
  int cl = lane & 15, rb = (lane >> 4) * 8;
  if (n0 < DK) {
    int col = n0 + cl;
#pragma unroll
    for (int r = 0; r < 8; ++r) Kall[(size_t)(m0 + rb + r) * DK + col] = acc[r];
  } else {
    int col = n0 - DK + cl;
#pragma unroll
    for (int r = 0; r < 8; ++r) Vall[(size_t)(m0 + rb + r) * VPAD + col] = (__bf16)acc[r];
  }
}

// ---------------- per-step kernels ----------------
// Step attention-in: q = h_b,n @ Wq ; z = q.k ; att1 = sigmoid(z/8); top-4 mask;
// build bf16 A-panel  [att1*V | h]  for the gates GEMM.   grid = BS, block = 256 (wave per block n)
__global__ void kStepAtt(int t, const float* __restrict__ hbuf, const float* __restrict__ WqT,
                         const float* __restrict__ Kall, const __bf16* __restrict__ Vall,
                         __bf16* __restrict__ Abuf, float* __restrict__ maskb) {
  __shared__ float zlds[NB];
  int b = blockIdx.x;
  int g = threadIdx.x >> 5;     // block n (one wave per block)
  int lane = threadIdx.x & 31;
  const float* hrow = hbuf + (size_t)b * NHID + g * BSZ;   // wave-uniform
  const float* w1 = WqT + (size_t)lane * BSZ;
  const float* w2 = WqT + (size_t)(lane + 32) * BSZ;
  float qa = 0.f, qb = 0.f;
  for (int k = 0; k < BSZ; ++k) { float hv = hrow[k]; qa += hv * w1[k]; qb += hv * w2[k]; }
  const float* kr = Kall + ((size_t)t * BS + b) * DK;
  float z = qa * kr[lane] + qb * kr[lane + 32];
  for (int m = 16; m > 0; m >>= 1) z += __shfl_xor(z, m, 32);
  if (lane == 0) zlds[g] = z;
  __syncthreads();
  float zg = zlds[g];
  int rank = 0;
  for (int m = 0; m < NB; ++m) {
    float zm = zlds[m];
    rank += (zm > zg) || (zm == zg && m < g);   // top_k tie-break: lower index wins
  }
  float msk  = (rank < TOPK) ? 1.f : 0.f;
  float att1 = sigf(zg * 0.125f);               // softmax([0,z']) -> sigmoid
  if (lane == 0) maskb[b * NB + g] = msk;
  const __bf16* vrow = Vall + ((size_t)t * BS + b) * VPAD;
  __bf16* arow = Abuf + ((size_t)g * BS + b) * KCAT;
  for (int c = lane; c < KCAT; c += 32) {
    float val = (c < VPAD) ? att1 * (float)vrow[c] : hrow[c - VPAD];
    arow[c] = (__bf16)val;
  }
}

// gates = A @ Wcat + b_lstm   per block n: (64 x 608)(608 x 1024).
// 512 waves: each computes a 16x64 strip (4 accumulators, A reused x4).
__global__ void kGates(const __bf16* __restrict__ Abuf, const __bf16* __restrict__ WcatT,
                       const float* __restrict__ b_lstm, float* __restrict__ gates) {
  int wid = blockIdx.x * 8 + (threadIdx.x >> 5);
  int lane = threadIdx.x & 31;
  int nblk = wid >> 6;
  int rem  = wid & 63;
  int m0 = (rem >> 4) * 16;
  int n0 = (rem & 15) * 64;
  const __bf16* A  = Abuf  + (size_t)nblk * BS * KCAT + (size_t)m0 * KCAT;
  const __bf16* Bb = WcatT + (size_t)nblk * 1024 * KCAT;
  v8f acc[4] = {};
  for (int k0 = 0; k0 < KCAT; k0 += 32) {
    v16bf a = frag_a_bf16(A, KCAT, k0, lane);
#pragma unroll
    for (int j = 0; j < 4; ++j) {
      const __bf16* bt = Bb + (size_t)(n0 + j * 16) * KCAT;
      __builtin_prefetch(bt + (size_t)(lane & 15) * KCAT + k0 + 64, 0, 1);
      v16bf bm = frag_b_bf16(bt, KCAT, k0, lane);
      acc[j] = WMMA_BF16(a, bm, acc[j]);
    }
  }
  int cl = lane & 15, rb = (lane >> 4) * 8;
#pragma unroll
  for (int j = 0; j < 4; ++j) {
    int col = n0 + j * 16 + cl;
    float bias = b_lstm[nblk * 1024 + col];
#pragma unroll
    for (int r = 0; r < 8; ++r)
      gates[((size_t)nblk * BS + m0 + rb + r) * 1024 + col] = acc[j][r] + bias;
  }
}

// LSTM pointwise -> c_new, h_new (f32) and bf16 h_new panel for the qkv GEMM.
__global__ void kLstm(const float* __restrict__ gates, const float* __restrict__ cbuf,
                      float* __restrict__ cnew, float* __restrict__ hnew,
                      __bf16* __restrict__ Hbf) {
  int idx = blockIdx.x * 256 + threadIdx.x;
  if (idx >= BS * NHID) return;
  int b = idx >> 11, r = idx & 2047, n = r >> 8, j = r & 255;
  const float* gp = gates + (size_t)(n * BS + b) * 1024;
  float ig = gp[j], fg = gp[256 + j], gg = gp[512 + j], og = gp[768 + j];
  float cn = sigf(fg) * cbuf[idx] + sigf(ig) * tanhf(gg);
  float hn = sigf(og) * tanhf(cn);
  cnew[idx] = cn; hnew[idx] = hn;
  Hbf[(size_t)(b * NB + n) * BSZ + j] = (__bf16)hn;
}

// qkv = h_new @ [Wq_c|Wk_c|Wv_c]   (M=512, K=256, N=384)
__global__ void kQkv(const __bf16* __restrict__ Hbf, const __bf16* __restrict__ WqkvT,
                     float* __restrict__ qkv) {
  int wid = blockIdx.x * 8 + (threadIdx.x >> 5);
  int lane = threadIdx.x & 31;
  int mt = wid / 24, nt = wid - mt * 24;
  int m0 = mt * 16, n0 = nt * 16;
  const __bf16* A  = Hbf   + (size_t)m0 * BSZ;
  const __bf16* BT = WqkvT + (size_t)n0 * BSZ;
  v8f acc = {};
  for (int k0 = 0; k0 < BSZ; k0 += 32) {
    v16bf a = frag_a_bf16(A, BSZ, k0, lane);
    v16bf b = frag_b_bf16(BT, BSZ, k0, lane);
    acc = WMMA_BF16(a, b, acc);
  }
  int col = n0 + (lane & 15), rb = (lane >> 4) * 8;
#pragma unroll
  for (int r = 0; r < 8; ++r)
    qkv[(size_t)(m0 + rb + r) * QKV_N + col] = acc[r];
}

// Cross-block 8x8 attention, oc @ Wo_c, masked state update, out[t] store.  grid = BS.
__global__ void kStepOut(int t, const float* __restrict__ qkv, const float* __restrict__ Wo_c,
                         const float* __restrict__ maskb, const float* __restrict__ hnew,
                         const float* __restrict__ cnew, float* __restrict__ hbuf,
                         float* __restrict__ cbuf, float* __restrict__ out) {
  __shared__ float qlds[NB * QKV_N];   // [n][q|k|v]
  __shared__ float oclds[NB * 128];
  int b = blockIdx.x, tid = threadIdx.x, lane = tid & 31;
  for (int i = tid; i < NB * QKV_N; i += 256)
    qlds[i] = qkv[(size_t)b * NB * QKV_N + i];
  __syncthreads();
  // teams of 8 lanes: (n,h) per team, lane-in-team = m
  int t0 = tid >> 3, n = t0 >> 2, h = t0 & 3, m = tid & 7;
  float s = 0.f;
  for (int k = 0; k < DK_C; ++k)
    s += qlds[n * QKV_N + h * 32 + k] * qlds[m * QKV_N + 128 + h * 32 + k];
  float mx = s;
  for (int d = 4; d > 0; d >>= 1) { float o = __shfl_xor(mx, d, 32); mx = o > mx ? o : mx; }
  float e = __expf((s - mx) * 0.17677669529663687f);   // 1/sqrt(32)
  float es = e;
  for (int d = 4; d > 0; d >>= 1) es += __shfl_xor(es, d, 32);
  float a = e / es;
  float acc[4] = {0.f, 0.f, 0.f, 0.f};
  int kb = (tid & 7) * 4;
  for (int m2 = 0; m2 < 8; ++m2) {
    float am = __shfl(a, (lane & ~7) + m2, 32);
#pragma unroll
    for (int kk = 0; kk < 4; ++kk)
      acc[kk] += am * qlds[m2 * QKV_N + 256 + h * 32 + kb + kk];
  }
#pragma unroll
  for (int kk = 0; kk < 4; ++kk) oclds[n * 128 + h * 32 + kb + kk] = acc[kk];
  __syncthreads();
  for (int p = tid; p < NB * BSZ; p += 256) {
    int nn = p >> 8, j = p & 255;
    const float* ocr = oclds + nn * 128;
    float delta = 0.f;
    for (int k = 0; k < 128; ++k) delta += ocr[k] * Wo_c[k * BSZ + j];
    float mk = maskb[b * NB + nn];
    int idx = b * NHID + nn * BSZ + j;
    float hx = (mk > 0.5f) ? (hnew[idx] + delta) : hbuf[idx];
    float cx = (mk > 0.5f) ? cnew[idx] : cbuf[idx];
    hbuf[idx] = hx; cbuf[idx] = cx;
    out[(size_t)t * BS * NHID + idx] = hx;
  }
}

// ---------------- host driver ----------------
extern "C" void kernel_launch(void* const* d_in, const int* in_sizes, int n_in,
                              void* d_out, int out_size, void* d_ws, size_t ws_size,
                              hipStream_t stream) {
  (void)in_sizes; (void)n_in; (void)out_size; (void)ws_size;
  const float* x     = (const float*)d_in[0];
  const float* h0    = (const float*)d_in[1];
  const float* c0    = (const float*)d_in[2];
  const float* W_enc = (const float*)d_in[3];
  const float* b_enc = (const float*)d_in[4];
  const float* Wq    = (const float*)d_in[5];
  const float* Wk    = (const float*)d_in[6];
  const float* Wv    = (const float*)d_in[7];
  const float* Wi    = (const float*)d_in[8];
  const float* Wh    = (const float*)d_in[9];
  const float* b_lstm= (const float*)d_in[10];
  const float* Wq_c  = (const float*)d_in[11];
  const float* Wk_c  = (const float*)d_in[12];
  const float* Wv_c  = (const float*)d_in[13];
  const float* Wo_c  = (const float*)d_in[14];
  float* out = (float*)d_out;

  char* ws = (char*)d_ws;
  size_t off = 0;
  auto alloc = [&](size_t bytes) -> void* {
    void* p = ws + off;
    off = (off + bytes + 255) & ~(size_t)255;
    return p;
  };
  __bf16* WencT = (__bf16*)alloc((size_t)512 * 512 * 2);
  __bf16* WkT   = (__bf16*)alloc((size_t)64 * 512 * 2);
  __bf16* WvT   = (__bf16*)alloc((size_t)VPAD * 512 * 2);
  float*  WqT   = (float*)alloc((size_t)64 * 256 * 4);
  __bf16* WqkvT = (__bf16*)alloc((size_t)QKV_N * 256 * 2);
  __bf16* WcatT = (__bf16*)alloc((size_t)NB * 1024 * KCAT * 2);
  __bf16* emb   = (__bf16*)alloc((size_t)T_STEPS * BS * NINP * 2);
  float*  Kall  = (float*)alloc((size_t)T_STEPS * BS * DK * 4);
  __bf16* Vall  = (__bf16*)alloc((size_t)T_STEPS * BS * VPAD * 2);
  __bf16* Abuf  = (__bf16*)alloc((size_t)NB * BS * KCAT * 2);
  float*  gates = (float*)alloc((size_t)NB * BS * 1024 * 4);
  float*  hbuf  = (float*)alloc((size_t)BS * NHID * 4);
  float*  cbuf  = (float*)alloc((size_t)BS * NHID * 4);
  float*  hnew  = (float*)alloc((size_t)BS * NHID * 4);
  float*  cnew  = (float*)alloc((size_t)BS * NHID * 4);
  __bf16* Hbf   = (__bf16*)alloc((size_t)BS * NHID * 2);
  float*  qkv   = (float*)alloc((size_t)BS * NB * QKV_N * 4);
  float*  maskb = (float*)alloc((size_t)BS * NB * 4);

  // setup
  kPrepWcat<<<(NB * 1024 * KCAT + 255) / 256, 256, 0, stream>>>(Wi, Wh, WcatT);
  kPrepMisc<<<(589824 + 255) / 256, 256, 0, stream>>>(W_enc, Wk, Wv, Wq, Wq_c, Wk_c, Wv_c,
                                                      WencT, WkT, WvT, WqT, WqkvT);
  kInitState<<<(BS * NHID + 255) / 256, 256, 0, stream>>>(h0, c0, hbuf, cbuf);
  kEmb<<<4096, 256, 0, stream>>>(x, WencT, b_enc, emb);     // 32768 tiles / 8 waves
  kKV<<<3328, 256, 0, stream>>>(emb, WkT, WvT, Kall, Vall); // 26624 tiles / 8 waves

  // sequential scan
  for (int t = 0; t < T_STEPS; ++t) {
    kStepAtt<<<BS, 256, 0, stream>>>(t, hbuf, WqT, Kall, Vall, Abuf, maskb);
    kGates<<<64, 256, 0, stream>>>(Abuf, WcatT, b_lstm, gates);
    kLstm<<<(BS * NHID + 255) / 256, 256, 0, stream>>>(gates, cbuf, cnew, hnew, Hbf);
    kQkv<<<96, 256, 0, stream>>>(Hbf, WqkvT, qkv);
    kStepOut<<<BS, 256, 0, stream>>>(t, qkv, Wo_c, maskb, hnew, cnew, hbuf, cbuf, out);
  }
}
